// Discriminator2_77309411726
// MI455X (gfx1250) — matrix operand order repaired and checked by reading, not compiled, not run on hardware
//
#include <hip/hip_runtime.h>

#define DD 512
#define SS 4
#define WAVES 4

typedef __attribute__((ext_vector_type(16))) _Float16 v16h;
typedef __attribute__((ext_vector_type(8)))  float    v8f;
typedef __attribute__((ext_vector_type(4)))  float    v4f;

__device__ __forceinline__ void cvt4(v16h& d, int o, v4f s) {
  d[o+0] = (_Float16)s[0]; d[o+1] = (_Float16)s[1];
  d[o+2] = (_Float16)s[2]; d[o+3] = (_Float16)s[3];
}

// Pre-pack W (f32, row-major [D,D]) into f16 in this code's contiguous-k WMMA B
// register layout, chunk-major:  for chunk blk = nb*16 + c  (16-col block nb,
// 32-deep K slice c), lane L's v16h lives at  wp[blk*512 + L*16], holding
// B[k, j] = W[nb*16 + (L&15), c*32 + (L>>4)*16 + t],  t = 0..15.
__global__ __launch_bounds__(32)
void pack_w_kernel(const float* __restrict__ W, _Float16* __restrict__ wp) {
  const int blk  = blockIdx.x;            // 0 .. (DD/16)*16 - 1  (= 512)
  const int nb   = blk >> 4;
  const int c    = blk & 15;
  const int lane = threadIdx.x & 31;
  const int hi   = lane >> 4;
  const int l16  = lane & 15;
  const float* src = W + (size_t)(nb * 16 + l16) * DD + c * 32 + hi * 16;
  v4f f0 = *(const v4f*)(src);
  v4f f1 = *(const v4f*)(src + 4);
  v4f f2 = *(const v4f*)(src + 8);
  v4f f3 = *(const v4f*)(src + 12);
  v16h b; cvt4(b, 0, f0); cvt4(b, 4, f1); cvt4(b, 8, f2); cvt4(b, 12, f3);
  *(v16h*)(wp + (size_t)blk * 512 + lane * 16) = b;
}

// One wave handles a 16-row tile of hc0:
//   U(16x512) = hc_tile(16x512) @ W^T  via v_wmma_f32_16x16x32_f16
// then rowwise dots with h_pl and the S gathered hc rows, lane-reduced.
// PACKED=true  : B comes pre-swizzled f16 from d_ws (no cvt in hot loop)
// PACKED=false : B converted from f32 W in-loop (fallback if ws too small)
template <bool PACKED>
__global__ __launch_bounds__(WAVES * 32)
void disc2_score_kernel(const float* __restrict__ hc,
                        const float* __restrict__ hpl,
                        const int*   __restrict__ samp,
                        const float* __restrict__ W,
                        const _Float16* __restrict__ wpk,
                        const float* __restrict__ bptr,
                        float* __restrict__ out,
                        int N, int n_tiles)
{
  __shared__ _Float16 ldsA[WAVES * 16 * DD];   // 64 KB: per-wave 16KB A tile (f16)
  const int wave = threadIdx.x >> 5;
  const int lane = threadIdx.x & 31;
  const int tile = blockIdx.x * WAVES + wave;
  if (tile >= n_tiles) return;                 // wave-uniform: EXEC stays all-1s for WMMA
  const int rowbase = tile * 16;
  const int hi  = lane >> 4;                   // lane half (0/1)
  const int l16 = lane & 15;
  const float bias = bptr[0];
  _Float16* myA = ldsA + wave * (16 * DD);

  // ---- stage A tile (16 x 512 of h_c) to LDS, f16, pre-swizzled to WMMA A layout ----
  // A 16x32, contiguous k order used here (same as B): lanes 0-15 (row=l16) hold K {0..15}; lanes 16-31 hold {16..31}
  {
    int arow = rowbase + l16; if (arow > N - 1) arow = N - 1;
    const int koff = hi ? 16 : 0;
    const float* src = hc + (size_t)arow * DD;
    #pragma unroll 4
    for (int c = 0; c < 16; ++c) {
      const int kb = c * 32;
      v4f f0 = *(const v4f*)(src + kb + koff);
      v4f f1 = *(const v4f*)(src + kb + koff + 4);
      v4f f2 = *(const v4f*)(src + kb + koff + 8);
      v4f f3 = *(const v4f*)(src + kb + koff + 12);
      v16h a; cvt4(a, 0, f0); cvt4(a, 4, f1); cvt4(a, 8, f2); cvt4(a, 12, f3);
      *(v16h*)(myA + c * 512 + lane * 16) = a;   // per-chunk, per-lane 32B slot
    }
  }

  // ---- preload gather indices for this lane's 8 rows (m = r + 8*hi) ----
  int sidx[SS][8];
  #pragma unroll
  for (int s = 0; s < SS; ++s)
    #pragma unroll
    for (int r = 0; r < 8; ++r) {
      int n = rowbase + r + 8 * hi; if (n > N - 1) n = N - 1;
      sidx[s][r] = samp[(size_t)s * N + n];
    }

  float sc1p[8];
  float sc2p[SS][8];
  #pragma unroll
  for (int r = 0; r < 8; ++r) {
    sc1p[r] = 0.f;
    #pragma unroll
    for (int s = 0; s < SS; ++s) sc2p[s][r] = 0.f;
  }

  // ---- 32 column blocks of U; K=512 in 16 steps of 32 ----
  for (int nb = 0; nb < DD / 16; ++nb) {
    const int nbase = nb * 16;

    v8f acc0 = {}; v8f acc1 = {};
    if (PACKED) {
      // B pre-swizzled: chunk blk = nb*16 + c, per-lane contiguous 32B v16h.
      const _Float16* wb = wpk + (size_t)(nb * 16) * 512 + lane * 16;
      __builtin_prefetch(wb + 16 * 512, 0, 0);   // next column block of packed W
      #pragma unroll
      for (int c = 0; c < 16; c += 2) {
        v16h a0 = *(const v16h*)(myA + (c + 0) * 512 + lane * 16);
        v16h a1 = *(const v16h*)(myA + (c + 1) * 512 + lane * 16);
        v16h b0 = *(const v16h*)(wb + (c + 0) * 512);
        v16h b1 = *(const v16h*)(wb + (c + 1) * 512);
        acc0 = __builtin_amdgcn_wmma_f32_16x16x32_f16(false, a0, false, b0,
                                                      (short)0, acc0, false, false);
        acc1 = __builtin_amdgcn_wmma_f32_16x16x32_f16(false, a1, false, b1,
                                                      (short)0, acc1, false, false);
      }
    } else {
      // Fallback: convert W f32->f16 in-loop. B[k,j] = W[nbase+j, k].
      const float* wrow = W + (size_t)(nbase + l16) * DD + hi * 16;
      __builtin_prefetch(wrow + 16 * DD, 0, 0);
      #pragma unroll
      for (int c = 0; c < 16; c += 2) {
        v16h a0 = *(const v16h*)(myA + (c + 0) * 512 + lane * 16);
        v16h a1 = *(const v16h*)(myA + (c + 1) * 512 + lane * 16);
        const float* wp0 = wrow + (c + 0) * 32;
        const float* wp1 = wrow + (c + 1) * 32;
        v16h b0, b1;
        cvt4(b0, 0, *(const v4f*)(wp0));      cvt4(b0, 4,  *(const v4f*)(wp0 + 4));
        cvt4(b0, 8, *(const v4f*)(wp0 + 8));  cvt4(b0, 12, *(const v4f*)(wp0 + 12));
        cvt4(b1, 0, *(const v4f*)(wp1));      cvt4(b1, 4,  *(const v4f*)(wp1 + 4));
        cvt4(b1, 8, *(const v4f*)(wp1 + 8));  cvt4(b1, 12, *(const v4f*)(wp1 + 12));
        acc0 = __builtin_amdgcn_wmma_f32_16x16x32_f16(false, a0, false, b0,
                                                      (short)0, acc0, false, false);
        acc1 = __builtin_amdgcn_wmma_f32_16x16x32_f16(false, a1, false, b1,
                                                      (short)0, acc1, false, false);
      }
    }

    // C/D layout: lane holds column j=l16 for rows m = r + 8*hi (r = vgpr index).
    #pragma unroll
    for (int r = 0; r < 8; ++r) {
      int n = rowbase + r + 8 * hi; if (n > N - 1) n = N - 1;
      const float u = acc0[r] + acc1[r];
      sc1p[r] += u * hpl[(size_t)n * DD + nbase + l16];
      #pragma unroll
      for (int s = 0; s < SS; ++s)
        sc2p[s][r] += u * hc[(size_t)sidx[s][r] * DD + nbase + l16];
    }
  }

  // ---- reduce each row's partial across its 16 column-lanes (xor tree stays in-half) ----
  #pragma unroll
  for (int r = 0; r < 8; ++r) {
    float v = sc1p[r];
    v += __shfl_xor(v, 8, 32);
    v += __shfl_xor(v, 4, 32);
    v += __shfl_xor(v, 2, 32);
    v += __shfl_xor(v, 1, 32);
    sc1p[r] = v;
    #pragma unroll
    for (int s = 0; s < SS; ++s) {
      float w = sc2p[s][r];
      w += __shfl_xor(w, 8, 32);
      w += __shfl_xor(w, 4, 32);
      w += __shfl_xor(w, 2, 32);
      w += __shfl_xor(w, 1, 32);
      sc2p[s][r] = w;
    }
  }

  // logits layout: [sc1 (N) | sc2 rows 0..S-1 | sc2 rows 0..S-1 again]
  if (l16 == 0) {
    #pragma unroll
    for (int r = 0; r < 8; ++r) {
      const int n = rowbase + r + 8 * hi;
      if (n < N) {
        out[n] = sc1p[r] + bias;
        #pragma unroll
        for (int s = 0; s < SS; ++s) {
          const float v = sc2p[s][r] + bias;
          out[(size_t)N * (1 + s) + n]      = v;
          out[(size_t)N * (1 + s + SS) + n] = v;
        }
      }
    }
  }
}

extern "C" void kernel_launch(void* const* d_in, const int* in_sizes, int n_in,
                              void* d_out, int out_size, void* d_ws, size_t ws_size,
                              hipStream_t stream) {
  const float* hc  = (const float*)d_in[0];   // h_c  [1,N,D] f32
  const float* hpl = (const float*)d_in[1];   // h_pl [1,N,D] f32
  const int*   smp = (const int*)  d_in[2];   // sample_list [S,N]
  const float* W   = (const float*)d_in[3];   // W [1,D,D] f32
  const float* b   = (const float*)d_in[4];   // bias [1]
  float* out = (float*)d_out;                 // [1, 9*N] f32

  const int N = in_sizes[0] / DD;             // 100000
  const int n_tiles = (N + 15) / 16;          // 6250
  dim3 grid((n_tiles + WAVES - 1) / WAVES);
  dim3 block(WAVES * 32);

  const size_t wpk_bytes = (size_t)DD * DD * sizeof(_Float16);  // 512 KB
  if (ws_size >= wpk_bytes) {
    _Float16* wpk = (_Float16*)d_ws;
    // 512 chunks, one wave each: pack W -> f16 B-layout (runs once per launch;
    // stream ordering serializes it ahead of the score kernel).
    hipLaunchKernelGGL(pack_w_kernel, dim3((DD / 16) * 16), dim3(32), 0, stream, W, wpk);
    hipLaunchKernelGGL((disc2_score_kernel<true>), grid, block, 0, stream,
                       hc, hpl, smp, W, wpk, b, out, N, n_tiles);
  } else {
    hipLaunchKernelGGL((disc2_score_kernel<false>), grid, block, 0, stream,
                       hc, hpl, smp, W, (const _Float16*)nullptr, b, out, N, n_tiles);
  }
}
